// Attention_11665131176370
// MI455X (gfx1250) — matrix operand order repaired and checked by reading, not compiled, not run on hardware
//
#include <hip/hip_runtime.h>
#include <hip/hip_fp16.h>

typedef __attribute__((ext_vector_type(16))) _Float16 v16h;
typedef __attribute__((ext_vector_type(8)))  _Float16 f16x8;
typedef __attribute__((ext_vector_type(2)))  _Float16 h2;
typedef __attribute__((ext_vector_type(8)))  float    v8f;
typedef __attribute__((ext_vector_type(4)))  float    f32x4;
typedef __attribute__((ext_vector_type(4)))  unsigned u32x4;
typedef __attribute__((ext_vector_type(4)))  int      i32x4;
typedef __attribute__((ext_vector_type(8)))  int      i32x8;

#define S_LEN 2048
#define D_DIM 64
#define QTILE 128      // q rows per block (8 waves x 16 rows)
#define WAVES 8
#define KTILE 32       // keys staged per iteration

#if defined(__AMDGCN__) && __has_builtin(__builtin_amdgcn_tensor_load_to_lds)
#define HAVE_TDM 1
#else
#define HAVE_TDM 0
#endif

#if HAVE_TDM
// One TDM 2D tile load: KTILE rows x 64 f32, row stride 64 elements.
// D# per CDNA5 ISA ch.8 (group0: count/lds_addr/global_addr/type,
// group1: data_size/dims/tile/strides; groups 2-3 unused for 2D).
__device__ __forceinline__ void tdm_load_tile(const float* gptr, unsigned lds_off) {
    unsigned long long ga = (unsigned long long)(uintptr_t)gptr;
    u32x4 g0;
    g0.x = 1u;                                       // count=1 (valid)
    g0.y = lds_off;                                  // lds_addr
    g0.z = (unsigned)(ga & 0xFFFFFFFFull);           // global_addr[31:0]
    g0.w = (unsigned)((ga >> 32) & 0x1FFFFFFull)     // global_addr[56:32]
         | (2u << 30);                               // type=2 ("image")
    i32x8 g1;
    g1[0] = (2 << 16);                 // data_size=2 (4B), mask=0, no flags
    g1[1] = (int)(64u << 16);          // tensor_dim0=64
    g1[2] = (int)(2048u << 16);        // tensor_dim1=2048 (lo16)
    g1[3] = (int)(64u << 16);          // tile_dim0=64
    g1[4] = KTILE;                     // tile_dim1=32, tile_dim2=0
    g1[5] = 64;                        // tensor_dim0_stride=64
    g1[6] = 0;
    g1[7] = 0;
    i32x4 gz = {0, 0, 0, 0};
#if __clang_major__ >= 23
    i32x8 gz8 = {0, 0, 0, 0, 0, 0, 0, 0};
    __builtin_amdgcn_tensor_load_to_lds(g0, g1, gz, gz, gz8, 0);
#else
    __builtin_amdgcn_tensor_load_to_lds(g0, g1, gz, gz, 0);
#endif
}
#endif

// 32 contiguous bytes of f16 from LDS -> 16-half fragment (2x ds_load_b128)
__device__ __forceinline__ v16h ld16h(const _Float16* p) {
    const f16x8* q = (const f16x8*)p;
    f16x8 lo = q[0], hi = q[1];
    return __builtin_shufflevector(lo, hi, 0,1,2,3,4,5,6,7,8,9,10,11,12,13,14,15);
}

__launch_bounds__(256)
__global__ void fa_wmma_kernel(const float* __restrict__ Q,
                               const float* __restrict__ K,
                               const float* __restrict__ V,
                               float* __restrict__ O)
{
#if HAVE_TDM
    __shared__ __align__(16) float k32[2][KTILE][D_DIM];     // TDM f32 staging, 16 KB
    __shared__ __align__(16) float v32[2][KTILE][D_DIM];     // 16 KB
#endif
    __shared__ __align__(16) _Float16 k16[KTILE][D_DIM];     // f16 K tile, 4 KB
    __shared__ __align__(16) _Float16 v16t[D_DIM][KTILE];    // f16 V^T tile, 4 KB
    __shared__ __align__(16) _Float16 p_lds[WAVES * 16 * KTILE]; // 8 KB P bounce

    const int lane   = threadIdx.x & 31;
    const int wave   = threadIdx.x >> 5;
    const int half16 = lane >> 4;
    const int l16    = lane & 15;

    const int head  = blockIdx.x >> 4;            // 64 (b,h) heads
    const int qtile = blockIdx.x & 15;
    const size_t hb = (size_t)head * S_LEN * D_DIM;
    const int q0    = qtile * QTILE + wave * 16;

    const float SCALE = 0.3535533905932738f;      // 64^-0.25 folded into Q

    // ---- Q -> two 16x32 f16 A fragments, pre-scaled -------------------------
    v16h a0, a1;
    {
        const float* qrow = Q + hb + (size_t)(q0 + l16) * D_DIM;
        const int d0 = 16 * half16;
        #pragma unroll
        for (int e = 0; e < 8; ++e) {
            a0[e]     = (_Float16)(qrow[d0 + e]      * SCALE);
            a0[8 + e] = (_Float16)(qrow[8 + d0 + e] * SCALE);
            a1[e]     = (_Float16)(qrow[32 + d0 + e] * SCALE);
            a1[8 + e] = (_Float16)(qrow[40 + d0 + e] * SCALE);
        }
    }

    v16h ones;                                    // B of all-ones: l = P @ 1
    #pragma unroll
    for (int e = 0; e < 16; ++e) ones[e] = (_Float16)1.0f;

    float m_r[8];
    v8f acc[4], lacc;
    const v8f vzero = {};
    #pragma unroll
    for (int j = 0; j < 8; ++j) m_r[j] = -3.0e38f;
    #pragma unroll
    for (int c = 0; c < 4; ++c) acc[c] = vzero;
    lacc = vzero;

    _Float16* pw = p_lds + wave * (16 * KTILE);

#if HAVE_TDM
    const unsigned koff[2] = { (unsigned)(uintptr_t)&k32[0][0][0],
                               (unsigned)(uintptr_t)&k32[1][0][0] };
    const unsigned voff[2] = { (unsigned)(uintptr_t)&v32[0][0][0],
                               (unsigned)(uintptr_t)&v32[1][0][0] };
    if (wave == 0) tdm_load_tile(K + hb, koff[0]);   // prologue DMA of tile 0
    if (wave == 1) tdm_load_tile(V + hb, voff[0]);
#endif

    for (int kb = 0; kb < S_LEN; kb += KTILE) {
        const int buf = (kb / KTILE) & 1;
        const int t   = threadIdx.x;
        const int key = t >> 3;            // staging work split: 8 floats/thread
        const int d0  = (t & 7) * 8;

#if HAVE_TDM
        if (kb + KTILE < S_LEN) {          // DMA tile kb+KTILE into other buffer
            if (wave == 0) tdm_load_tile(K + hb + (size_t)(kb + KTILE) * D_DIM, koff[buf ^ 1]);
            if (wave == 1) tdm_load_tile(V + hb + (size_t)(kb + KTILE) * D_DIM, voff[buf ^ 1]);
            if (wave < 2) __builtin_amdgcn_s_wait_tensorcnt(1);   // tile kb landed
        } else {
            if (wave < 2) __builtin_amdgcn_s_wait_tensorcnt(0);
        }
        __syncthreads();   // barrier A: f32 tile ready AND prior compute finished

        // cooperative f32 -> f16 conversion (once per tile, shared by 8 waves)
        {
            const f32x4* sk = (const f32x4*)&k32[buf][0][0];
            f32x4 x0 = sk[2 * t], x1 = sk[2 * t + 1];
            f16x8 hk;
            hk[0] = (_Float16)x0.x; hk[1] = (_Float16)x0.y;
            hk[2] = (_Float16)x0.z; hk[3] = (_Float16)x0.w;
            hk[4] = (_Float16)x1.x; hk[5] = (_Float16)x1.y;
            hk[6] = (_Float16)x1.z; hk[7] = (_Float16)x1.w;
            ((f16x8*)&k16[0][0])[t] = hk;

            const f32x4* sv = (const f32x4*)&v32[buf][key][d0];
            f32x4 y0 = sv[0], y1 = sv[1];
            v16t[d0 + 0][key] = (_Float16)y0.x;  v16t[d0 + 1][key] = (_Float16)y0.y;
            v16t[d0 + 2][key] = (_Float16)y0.z;  v16t[d0 + 3][key] = (_Float16)y0.w;
            v16t[d0 + 4][key] = (_Float16)y1.x;  v16t[d0 + 5][key] = (_Float16)y1.y;
            v16t[d0 + 6][key] = (_Float16)y1.z;  v16t[d0 + 7][key] = (_Float16)y1.w;
        }
        __syncthreads();   // barrier B: f16 tiles published
#else
        __syncthreads();   // prior compute finished before overwrite
        {
            const f32x4* gk = (const f32x4*)(K + hb + (size_t)kb * D_DIM);
            f32x4 x0 = gk[2 * t], x1 = gk[2 * t + 1];
            f16x8 hk;
            hk[0] = (_Float16)x0.x; hk[1] = (_Float16)x0.y;
            hk[2] = (_Float16)x0.z; hk[3] = (_Float16)x0.w;
            hk[4] = (_Float16)x1.x; hk[5] = (_Float16)x1.y;
            hk[6] = (_Float16)x1.z; hk[7] = (_Float16)x1.w;
            ((f16x8*)&k16[0][0])[t] = hk;

            const f32x4* gv = (const f32x4*)(V + hb + (size_t)(kb + key) * D_DIM + d0);
            f32x4 y0 = gv[0], y1 = gv[1];
            v16t[d0 + 0][key] = (_Float16)y0.x;  v16t[d0 + 1][key] = (_Float16)y0.y;
            v16t[d0 + 2][key] = (_Float16)y0.z;  v16t[d0 + 3][key] = (_Float16)y0.w;
            v16t[d0 + 4][key] = (_Float16)y1.x;  v16t[d0 + 5][key] = (_Float16)y1.y;
            v16t[d0 + 6][key] = (_Float16)y1.z;  v16t[d0 + 7][key] = (_Float16)y1.w;
        }
        __syncthreads();
#endif

        // ---------- scores: S = Q @ K^T (two 16x16 blocks, f16 frags from LDS)
        v8f c0 = vzero, c1 = vzero;
        {
            v16h b00 = ld16h(&k16[l16][16 * half16]);
            v16h b01 = ld16h(&k16[l16][32 + 16 * half16]);
            v16h b10 = ld16h(&k16[16 + l16][16 * half16]);
            v16h b11 = ld16h(&k16[16 + l16][32 + 16 * half16]);
            c0 = __builtin_amdgcn_wmma_f32_16x16x32_f16(false, a0, false, b00, (short)0, c0, false, false);
            c0 = __builtin_amdgcn_wmma_f32_16x16x32_f16(false, a1, false, b01, (short)0, c0, false, false);
            c1 = __builtin_amdgcn_wmma_f32_16x16x32_f16(false, a0, false, b10, (short)0, c1, false, false);
            c1 = __builtin_amdgcn_wmma_f32_16x16x32_f16(false, a1, false, b11, (short)0, c1, false, false);
        }

        // ---------- online softmax: packed-f16 max butterfly (v_pk_max_num_f16)
        // Any stabilization shift m works as long as exp doesn't overflow; the
        // identical shifted P feeds both O and l, so rounding of m cancels.
        float alpha[8];
        {
            float tm[8];
            #pragma unroll
            for (int j = 0; j < 8; ++j) tm[j] = fmaxf(c0[j], c1[j]);
            h2 pk[4];
            #pragma unroll
            for (int p = 0; p < 4; ++p) {
                pk[p].x = (_Float16)tm[2 * p];
                pk[p].y = (_Float16)tm[2 * p + 1];
            }
            #pragma unroll
            for (int m = 1; m <= 8; m <<= 1) {       // butterfly within 16 lanes
                #pragma unroll
                for (int p = 0; p < 4; ++p) {
                    int x = __shfl_xor(__builtin_bit_cast(int, pk[p]), m, 32);
                    pk[p] = __builtin_elementwise_max(pk[p], __builtin_bit_cast(h2, x));
                }
            }
            #pragma unroll
            for (int j = 0; j < 8; ++j) {
                const float tmax = (float)((j & 1) ? pk[j >> 1].y : pk[j >> 1].x);
                const float mn = fmaxf(m_r[j], tmax);
                alpha[j] = __expf(m_r[j] - mn);
                m_r[j]   = mn;
                c0[j] -= mn;     // keep (s - m); exp applied after re-layout
                c1[j] -= mn;
            }
        }

        // ---------- (s-m): C layout -> LDS -> A fragment, then exp in f16 -----
        #pragma unroll
        for (int j = 0; j < 8; ++j) {
            const int row = j + 8 * half16;
            pw[row * KTILE + l16]      = (_Float16)c0[j];
            pw[row * KTILE + 16 + l16] = (_Float16)c1[j];
        }
        v16h pa;   // same-wave DS in-order; wave-private region, no barrier
        {
            union HU { f16x8 v; __half2 h[4]; };
            const f16x8* q = (const f16x8*)(pw + l16 * KTILE + 16 * half16);
            HU lo, hi;
            lo.v = q[0];  hi.v = q[1];
            #pragma unroll
            for (int i = 0; i < 4; ++i) {        // packed v_exp_f16 (TRANS)
                lo.h[i] = h2exp(lo.h[i]);
                hi.h[i] = h2exp(hi.h[i]);
            }
            pa = __builtin_shufflevector(lo.v, hi.v,
                                         0,1,2,3,4,5,6,7,8,9,10,11,12,13,14,15);
        }

        // ---------- l += P @ 1 and O += P @ V (5 WMMAs) -----------------------
        #pragma unroll
        for (int j = 0; j < 8; ++j) lacc[j] *= alpha[j];
        lacc = __builtin_amdgcn_wmma_f32_16x16x32_f16(false, pa, false, ones, (short)0, lacc, false, false);

        #pragma unroll
        for (int c = 0; c < 4; ++c) {
            v16h vb = ld16h(&v16t[c * 16 + l16][16 * half16]);   // N=d, K=key
            #pragma unroll
            for (int j = 0; j < 8; ++j)
                acc[c][j] *= alpha[j];
            acc[c] = __builtin_amdgcn_wmma_f32_16x16x32_f16(false, pa, false, vb, (short)0, acc[c], false, false);
        }
    }

    // ---------- epilogue: multiply by 1/rowsum, coalesced fp32 store ----------
    float inv[8];
    #pragma unroll
    for (int j = 0; j < 8; ++j) inv[j] = 1.0f / lacc[j];
    #pragma unroll
    for (int c = 0; c < 4; ++c) {
        #pragma unroll
        for (int j = 0; j < 8; ++j) {
            const int row = j + 8 * half16;
            O[hb + (size_t)(q0 + row) * D_DIM + c * 16 + l16] = acc[c][j] * inv[j];
        }
    }
}

extern "C" void kernel_launch(void* const* d_in, const int* in_sizes, int n_in,
                              void* d_out, int out_size, void* d_ws, size_t ws_size,
                              hipStream_t stream) {
    const float* Q = (const float*)d_in[0];
    const float* K = (const float*)d_in[1];
    const float* V = (const float*)d_in[2];
    float* O = (float*)d_out;
    (void)in_sizes; (void)n_in; (void)out_size; (void)d_ws; (void)ws_size;

    dim3 grid(64 * (S_LEN / QTILE));   // same-head blocks adjacent -> K/V L2 reuse
    dim3 block(256);
    fa_wmma_kernel<<<grid, block, 0, stream>>>(Q, K, V, O);
}
